// MultiHeadAttention_8074538516545
// MI455X (gfx1250) — matrix operand order repaired and checked by reading, not compiled, not run on hardware
//
#include <hip/hip_runtime.h>

// ---------------------------------------------------------------------------
// MI455X (gfx1250) multi-head attention forward.
// GEMMs on v_wmma_f32_16x16x32_f16 (fp32 accumulate); tile staging via the
// Tensor Data Mover (tensor_load_to_lds + s_wait_tensorcnt) when available.
// B=2, N=2048, DIM=1024, H=16, d_head=64, scale = DIM^-0.5 = 1/32.
// ---------------------------------------------------------------------------

typedef __attribute__((ext_vector_type(16))) _Float16 v16h;
typedef __attribute__((ext_vector_type(8)))  float    v8f;

#define NB   2
#define NSEQ 2048
#define DIM  1024
#define NH   16
#define DH   64
#define MTOT (NB * NSEQ)   // 4096

#if defined(__has_builtin)
#if __has_builtin(__builtin_amdgcn_tensor_load_to_lds) && \
    __has_builtin(__builtin_amdgcn_s_wait_tensorcnt)
#define USE_TDM 1
#endif
#endif
#ifndef USE_TDM
#define USE_TDM 0
#endif

static __device__ __forceinline__ v8f wmma16(v16h a, v16h b, v8f c) {
  return __builtin_amdgcn_wmma_f32_16x16x32_f16(false, a, false, b, (short)0, c,
                                                false, false);
}

// A-fragment: 16x32 f16 tile, element (row, k) at src[row*ld + k].
// Contiguous k order used here (A and B match): lanes 0-15 -> row=lane, K=0..15; lanes 16-31 -> K=16..31.
static __device__ __forceinline__ v16h load_frag_a(const _Float16* src, int ld) {
  const unsigned lane = threadIdx.x & 31u;
  const unsigned row  = lane & 15u;
  const unsigned kb   = (lane < 16u) ? 0u : 16u;
  const _Float16* p = src + (size_t)row * ld + kb;
  v16h a;
#pragma unroll
  for (int i = 0; i < 8; ++i) a[i] = p[i];
#pragma unroll
  for (int i = 0; i < 8; ++i) a[8 + i] = p[8 + i];
  return a;
}

// B-fragment: 32(K)x16(N) tile, element (col n, k) at src[n*ld + k].
// Contiguous k order used here (A and B match): lanes 0-15 -> col=lane, K=0..15; lanes 16-31 -> col=lane-16, K=16..31.
static __device__ __forceinline__ v16h load_frag_b(const _Float16* src, int ld) {
  const unsigned lane = threadIdx.x & 31u;
  const unsigned col  = lane & 15u;
  const unsigned kb   = (lane < 16u) ? 0u : 16u;
  const _Float16* p = src + (size_t)col * ld + kb;
  v16h b;
#pragma unroll
  for (int i = 0; i < 16; ++i) b[i] = p[i];
  return b;
}

// ---------------------------------------------------------------------------
// Tensor Data Mover: 2D tile load (global -> LDS), descriptor per ISA ch.8.
// Element size fixed at 4 bytes; dims/strides given in dwords.
// tile_d1 == 0 -> 1D copy of tile_d0 dwords.
// ---------------------------------------------------------------------------
#if USE_TDM
typedef __attribute__((ext_vector_type(4))) unsigned int u32x4;
typedef __attribute__((ext_vector_type(8))) int i32x8;
typedef __attribute__((ext_vector_type(4))) int i32x4;

static __device__ __forceinline__ uint32_t lds_offset(const void* p) {
  // Generic pointers into LDS carry the byte offset in the low 32 bits.
  return (uint32_t)(uintptr_t)p;
}

static __device__ __forceinline__ void tdm_load_2d(
    uint32_t lds, const void* gptr, uint32_t tile_d0_dw, uint32_t tile_d1,
    uint32_t tensor_d0_dw, uint32_t tensor_d1, uint32_t stride0_dw) {
  const uint64_t ga = (uint64_t)(uintptr_t)gptr;
  u32x4 g0;
  g0[0] = 1u;                                              // count=1, user D#
  g0[1] = lds;                                             // lds_addr
  g0[2] = (uint32_t)ga;                                    // global_addr[31:0]
  g0[3] = (uint32_t)((ga >> 32) & 0x1FFFFFFu) | (2u << 30);// addr[56:32] | type=2
  i32x8 g1;
  g1[0] = (int)(2u << 16);                                 // data_size=4B
  g1[1] = (int)((tensor_d0_dw & 0xFFFFu) << 16);           // tensor_dim0 lo16
  g1[2] = (int)(((tensor_d0_dw >> 16) & 0xFFFFu) |
                ((tensor_d1 & 0xFFFFu) << 16));            // dim0 hi | dim1 lo
  g1[3] = (int)(((tensor_d1 >> 16) & 0xFFFFu) |
                ((tile_d0_dw & 0xFFFFu) << 16));           // dim1 hi | tile_dim0
  g1[4] = (int)(tile_d1 & 0xFFFFu);                        // tile_dim1, tile_dim2=0
  g1[5] = (int)stride0_dw;                                 // tensor_dim0_stride lo32
  g1[6] = 0;                                               // stride0 hi, stride1 lo
  g1[7] = 0;                                               // stride1 hi
  i32x4 z4 = {};
  i32x8 z8 = {};
  __builtin_amdgcn_tensor_load_to_lds(g0, g1, z4, z4, z8, 0);
}
#endif

// ---------------------------------------------------------------------------
// fp32 -> f16 conversion
// ---------------------------------------------------------------------------
__global__ void cvt_f16_kernel(const float* __restrict__ in,
                               _Float16* __restrict__ out, int n) {
  int i = blockIdx.x * blockDim.x + threadIdx.x;
  if (i < n) out[i] = (_Float16)in[i];
}

// ---------------------------------------------------------------------------
// QKV projection: C[m, nf] = sum_k X[m,k] * Wqkv[nf,k],  m<4096, nf<3072, k<1024
// Epilogue scatters into q[bh][n][d], k[bh][n][d], vT[bh][d][n] as f16.
// Block: 256 threads (8 waves), tile 64(M) x 128(N), K-step 32.
// ---------------------------------------------------------------------------
__global__ __launch_bounds__(256) void qkv_gemm_kernel(
    const _Float16* __restrict__ X, const _Float16* __restrict__ W,
    _Float16* __restrict__ Q, _Float16* __restrict__ K,
    _Float16* __restrict__ VT) {
  __shared__ _Float16 xs[64 * 32];    // 4 KB
  __shared__ _Float16 ws[128 * 32];   // 8 KB
  const int mt = blockIdx.x;          // 0..63
  const int nt = blockIdx.y;          // 0..23
  const int tid = threadIdx.x;
  const int wave = tid >> 5;
  const unsigned lane = tid & 31u;
  const int wr = wave >> 1;           // 0..3 (row group of 16)
  const int wc = wave & 1;            // 0..1 (col group of 64)
  const int m0 = mt * 64;
  const int n0 = nt * 128;
  v8f c[4] = {};

  for (int kk = 0; kk < DIM; kk += 32) {
#if USE_TDM
    if (wave == 0) {
      tdm_load_2d(lds_offset(xs), X + (size_t)m0 * DIM + kk, 16, 64, 16, 64, DIM / 2);
      tdm_load_2d(lds_offset(ws), W + (size_t)n0 * DIM + kk, 16, 128, 16, 128, DIM / 2);
      __builtin_amdgcn_s_wait_tensorcnt(0);
    }
#else
    {
      const uint32_t* xg = (const uint32_t*)(X + (size_t)m0 * DIM + kk);
      uint32_t* xsd = (uint32_t*)xs;
#pragma unroll
      for (int i = 0; i < 4; ++i) {
        int idx = tid + i * 256;
        xsd[idx] = xg[(size_t)(idx >> 4) * (DIM / 2) + (idx & 15)];
      }
      const uint32_t* wg = (const uint32_t*)(W + (size_t)n0 * DIM + kk);
      uint32_t* wsd = (uint32_t*)ws;
#pragma unroll
      for (int i = 0; i < 8; ++i) {
        int idx = tid + i * 256;
        wsd[idx] = wg[(size_t)(idx >> 4) * (DIM / 2) + (idx & 15)];
      }
    }
#endif
    __syncthreads();
    v16h a  = load_frag_a(xs + wr * 16 * 32, 32);
    v16h b0 = load_frag_b(ws + (wc * 64 + 0) * 32, 32);
    v16h b1 = load_frag_b(ws + (wc * 64 + 16) * 32, 32);
    v16h b2 = load_frag_b(ws + (wc * 64 + 32) * 32, 32);
    v16h b3 = load_frag_b(ws + (wc * 64 + 48) * 32, 32);
    c[0] = wmma16(a, b0, c[0]);
    c[1] = wmma16(a, b1, c[1]);
    c[2] = wmma16(a, b2, c[2]);
    c[3] = wmma16(a, b3, c[3]);
    __syncthreads();
  }

  const unsigned colL = lane & 15u;
  const unsigned hi = lane >> 4;
#pragma unroll
  for (int j = 0; j < 4; ++j) {
#pragma unroll
    for (int g = 0; g < 8; ++g) {
      int m  = m0 + wr * 16 + g + 8 * (int)hi;
      int nf = n0 + wc * 64 + j * 16 + (int)colL;
      _Float16 val = (_Float16)c[j][g];
      int which = nf >> 10;                 // 0=q 1=k 2=v
      int rem   = nf & 1023;
      int h     = rem >> 6;
      int dd    = rem & 63;
      int bb    = m >> 11;
      int n     = m & 2047;
      size_t bh = (size_t)(bb * NH + h);
      if (which == 0)      Q[(bh * NSEQ + n) * DH + dd] = val;
      else if (which == 1) K[(bh * NSEQ + n) * DH + dd] = val;
      else                 VT[(bh * DH + dd) * NSEQ + n] = val;
    }
  }
}

// ---------------------------------------------------------------------------
// Flash attention core. One block = 4 waves = 64 query rows of one (b,h).
// Streams 32 keys per iteration: S = Q(16x64) Kt -> online softmax -> O += P V.
// ---------------------------------------------------------------------------
__global__ __launch_bounds__(128) void attn_kernel(
    const _Float16* __restrict__ Q, const _Float16* __restrict__ K,
    const _Float16* __restrict__ VT, _Float16* __restrict__ AO) {
  __shared__ _Float16 ks[32 * 64];        // K tile  [key32][d64]  4 KB
  __shared__ _Float16 vs[64 * 32];        // Vt tile [d64][key32]  4 KB
  __shared__ _Float16 ps[4][16 * 32];     // per-wave P staging    4 KB

  const int bh  = blockIdx.x;             // 0..31
  const int rt  = blockIdx.y;             // 0..31
  const int tid = threadIdx.x;
  const int wave = tid >> 5;
  const unsigned lane = tid & 31u;
  const unsigned colL = lane & 15u;
  const unsigned hi   = lane >> 4;
  const int row0 = rt * 64 + wave * 16;
  const size_t base = (size_t)bh * NSEQ * DH;   // same for Q, K, VT slices

  // Q fragments for this wave's 16 rows (d 0..31 and 32..63), reused 64 times.
  v16h aq0 = load_frag_a(Q + base + (size_t)row0 * DH + 0, DH);
  v16h aq1 = load_frag_a(Q + base + (size_t)row0 * DH + 32, DH);

  v8f o[4] = {};
  float mrw[8], lrw[8];
#pragma unroll
  for (int g = 0; g < 8; ++g) { mrw[g] = -1.0e30f; lrw[g] = 0.0f; }
  const float scale = 0.03125f;           // 1024^-0.5

  for (int jb = 0; jb < NSEQ; jb += 32) {
#if USE_TDM
    if (wave == 0) {
      // K tile: one contiguous 4 KB span -> 1D copy of 1024 dwords.
      tdm_load_2d(lds_offset(ks), K + base + (size_t)jb * DH, 1024, 0, 1024, 1, 1024);
      // Vt tile: 64 rows x 16 dwords, row stride 1024 dwords.
      tdm_load_2d(lds_offset(vs), VT + base + jb, 16, 64, 16, 64, NSEQ / 2);
      __builtin_amdgcn_s_wait_tensorcnt(0);
    }
#else
    {
      const uint32_t* kg = (const uint32_t*)(K + base + (size_t)jb * DH);
      uint32_t* ksd = (uint32_t*)ks;
#pragma unroll
      for (int i = 0; i < 8; ++i) ksd[tid + i * 128] = kg[tid + i * 128];
      const uint32_t* vg = (const uint32_t*)(VT + base + jb);
      uint32_t* vsd = (uint32_t*)vs;
#pragma unroll
      for (int i = 0; i < 8; ++i) {
        int idx = tid + i * 128;
        vsd[idx] = vg[(size_t)(idx >> 4) * (NSEQ / 2) + (idx & 15)];
      }
    }
#endif
    __syncthreads();

    // Scores: S0 = keys jb..jb+15, S1 = keys jb+16..jb+31.
    v8f s0 = {}, s1 = {};
    v16h kb0 = load_frag_b(ks + 0 * 64 + 0, 64);
    v16h kb1 = load_frag_b(ks + 0 * 64 + 32, 64);
    v16h kb2 = load_frag_b(ks + 16 * 64 + 0, 64);
    v16h kb3 = load_frag_b(ks + 16 * 64 + 32, 64);
    s0 = wmma16(aq0, kb0, s0);
    s0 = wmma16(aq1, kb1, s0);
    s1 = wmma16(aq0, kb2, s1);
    s1 = wmma16(aq1, kb3, s1);

    // Online softmax, row = g + 8*hi spread over 16 lanes of a half-wave.
#pragma unroll
    for (int g = 0; g < 8; ++g) {
      float v0 = s0[g] * scale, v1 = s1[g] * scale;
      float tmax = fmaxf(v0, v1);
      tmax = fmaxf(tmax, __shfl_xor(tmax, 1));
      tmax = fmaxf(tmax, __shfl_xor(tmax, 2));
      tmax = fmaxf(tmax, __shfl_xor(tmax, 4));
      tmax = fmaxf(tmax, __shfl_xor(tmax, 8));
      float mnew = fmaxf(mrw[g], tmax);
      float corr = __expf(mrw[g] - mnew);
      float p0 = __expf(v0 - mnew);
      float p1 = __expf(v1 - mnew);
      float psum = p0 + p1;
      psum += __shfl_xor(psum, 1);
      psum += __shfl_xor(psum, 2);
      psum += __shfl_xor(psum, 4);
      psum += __shfl_xor(psum, 8);
      lrw[g] = lrw[g] * corr + psum;
      mrw[g] = mnew;
#pragma unroll
      for (int dc = 0; dc < 4; ++dc) o[dc][g] *= corr;
      _Float16* pp = ps[wave] + (size_t)(g + 8 * (int)hi) * 32;
      pp[colL]      = (_Float16)p0;
      pp[colL + 16] = (_Float16)p1;
    }
    __syncthreads();

    // O += P(16x32) x V(32x64)
    v16h ap  = load_frag_a(ps[wave], 32);
    v16h bv0 = load_frag_b(vs + 0 * 16 * 32, 32);
    v16h bv1 = load_frag_b(vs + 1 * 16 * 32, 32);
    v16h bv2 = load_frag_b(vs + 2 * 16 * 32, 32);
    v16h bv3 = load_frag_b(vs + 3 * 16 * 32, 32);
    o[0] = wmma16(ap, bv0, o[0]);
    o[1] = wmma16(ap, bv1, o[1]);
    o[2] = wmma16(ap, bv2, o[2]);
    o[3] = wmma16(ap, bv3, o[3]);
    __syncthreads();
  }

  // Normalize and store f16 attention output at [b, n, h*64 + dd].
  const int bb = bh >> 4, h = bh & 15;
#pragma unroll
  for (int dc = 0; dc < 4; ++dc) {
#pragma unroll
    for (int g = 0; g < 8; ++g) {
      int n  = row0 + g + 8 * (int)hi;
      int dd = dc * 16 + (int)colL;
      float val = o[dc][g] / lrw[g];
      AO[((size_t)(bb * NSEQ + n)) * DIM + h * DH + dd] = (_Float16)val;
    }
  }
}

// ---------------------------------------------------------------------------
// Output projection: OUT[m, nf] = sum_k AO[m,k]*Wout[nf,k] + bias[nf], fp32 out.
// ---------------------------------------------------------------------------
__global__ __launch_bounds__(256) void out_gemm_kernel(
    const _Float16* __restrict__ AO, const _Float16* __restrict__ W,
    const float* __restrict__ bias, float* __restrict__ OUT) {
  __shared__ _Float16 xs[64 * 32];
  __shared__ _Float16 ws[128 * 32];
  const int mt = blockIdx.x;          // 0..63
  const int nt = blockIdx.y;          // 0..7
  const int tid = threadIdx.x;
  const int wave = tid >> 5;
  const unsigned lane = tid & 31u;
  const int wr = wave >> 1;
  const int wc = wave & 1;
  const int m0 = mt * 64;
  const int n0 = nt * 128;
  v8f c[4] = {};

  for (int kk = 0; kk < DIM; kk += 32) {
#if USE_TDM
    if (wave == 0) {
      tdm_load_2d(lds_offset(xs), AO + (size_t)m0 * DIM + kk, 16, 64, 16, 64, DIM / 2);
      tdm_load_2d(lds_offset(ws), W + (size_t)n0 * DIM + kk, 16, 128, 16, 128, DIM / 2);
      __builtin_amdgcn_s_wait_tensorcnt(0);
    }
#else
    {
      const uint32_t* xg = (const uint32_t*)(AO + (size_t)m0 * DIM + kk);
      uint32_t* xsd = (uint32_t*)xs;
#pragma unroll
      for (int i = 0; i < 4; ++i) {
        int idx = tid + i * 256;
        xsd[idx] = xg[(size_t)(idx >> 4) * (DIM / 2) + (idx & 15)];
      }
      const uint32_t* wg = (const uint32_t*)(W + (size_t)n0 * DIM + kk);
      uint32_t* wsd = (uint32_t*)ws;
#pragma unroll
      for (int i = 0; i < 8; ++i) {
        int idx = tid + i * 256;
        wsd[idx] = wg[(size_t)(idx >> 4) * (DIM / 2) + (idx & 15)];
      }
    }
#endif
    __syncthreads();
    v16h a  = load_frag_a(xs + wr * 16 * 32, 32);
    v16h b0 = load_frag_b(ws + (wc * 64 + 0) * 32, 32);
    v16h b1 = load_frag_b(ws + (wc * 64 + 16) * 32, 32);
    v16h b2 = load_frag_b(ws + (wc * 64 + 32) * 32, 32);
    v16h b3 = load_frag_b(ws + (wc * 64 + 48) * 32, 32);
    c[0] = wmma16(a, b0, c[0]);
    c[1] = wmma16(a, b1, c[1]);
    c[2] = wmma16(a, b2, c[2]);
    c[3] = wmma16(a, b3, c[3]);
    __syncthreads();
  }

  const unsigned colL = lane & 15u;
  const unsigned hi = lane >> 4;
#pragma unroll
  for (int j = 0; j < 4; ++j) {
#pragma unroll
    for (int g = 0; g < 8; ++g) {
      int m  = m0 + wr * 16 + g + 8 * (int)hi;
      int nf = n0 + wc * 64 + j * 16 + (int)colL;
      OUT[(size_t)m * DIM + nf] = c[j][g] + bias[nf];
    }
  }
}

// ---------------------------------------------------------------------------
// Workspace layout (bytes): 48 MB total.
// ---------------------------------------------------------------------------
static constexpr size_t OFF_XH   = 0;                       // 4096*1024 f16 = 8 MB
static constexpr size_t OFF_WQKV = OFF_XH   + 8388608;      // 3072*1024 f16 = 6 MB
static constexpr size_t OFF_WOUT = OFF_WQKV + 6291456;      // 1024*1024 f16 = 2 MB
static constexpr size_t OFF_Q    = OFF_WOUT + 2097152;      // 32*2048*64 f16 = 8 MB
static constexpr size_t OFF_K    = OFF_Q    + 8388608;
static constexpr size_t OFF_VT   = OFF_K    + 8388608;
static constexpr size_t OFF_AO   = OFF_VT   + 8388608;      // 4096*1024 f16 = 8 MB

extern "C" void kernel_launch(void* const* d_in, const int* in_sizes, int n_in,
                              void* d_out, int out_size, void* d_ws, size_t ws_size,
                              hipStream_t stream) {
  (void)in_sizes; (void)n_in; (void)out_size; (void)ws_size;
  const float* x     = (const float*)d_in[0];
  const float* w_qkv = (const float*)d_in[1];
  const float* w_out = (const float*)d_in[2];
  const float* b_out = (const float*)d_in[3];
  float* out = (float*)d_out;
  char* ws = (char*)d_ws;

  _Float16* xh    = (_Float16*)(ws + OFF_XH);
  _Float16* wqkvh = (_Float16*)(ws + OFF_WQKV);
  _Float16* wouth = (_Float16*)(ws + OFF_WOUT);
  _Float16* qb    = (_Float16*)(ws + OFF_Q);
  _Float16* kb    = (_Float16*)(ws + OFF_K);
  _Float16* vtb   = (_Float16*)(ws + OFF_VT);
  _Float16* aob   = (_Float16*)(ws + OFF_AO);

  cvt_f16_kernel<<<(MTOT * DIM + 255) / 256, 256, 0, stream>>>(x, xh, MTOT * DIM);
  cvt_f16_kernel<<<(3 * DIM * DIM + 255) / 256, 256, 0, stream>>>(w_qkv, wqkvh, 3 * DIM * DIM);
  cvt_f16_kernel<<<(DIM * DIM + 255) / 256, 256, 0, stream>>>(w_out, wouth, DIM * DIM);

  qkv_gemm_kernel<<<dim3(MTOT / 64, (3 * DIM) / 128), 256, 0, stream>>>(
      xh, wqkvh, qb, kb, vtb);

  attn_kernel<<<dim3(NB * NH, NSEQ / 64), 128, 0, stream>>>(qb, kb, vtb, aob);

  out_gemm_kernel<<<dim3(MTOT / 64, DIM / 128), 256, 0, stream>>>(
      aob, wouth, b_out, out);
}
